// SymmetrizationNetMLP_283467842606
// MI455X (gfx1250) — compile-verified, matrix operand order checked by reading, not run on hardware
//
#include <hip/hip_runtime.h>

typedef __attribute__((ext_vector_type(16))) __bf16 v16bf;
typedef __attribute__((ext_vector_type(8)))  float  v8f;

// ---- workspace layout (bytes) ----
#define OFF_M    0          // 262144 f32  (per-pixel mean)           1 MiB
#define OFF_S0   1048576    // 1024 f32    (rowsum of W0)             4 KiB
#define OFF_W1P  1052672    // 1024x1024 bf16 packed W1^T tiles       2 MiB
#define OFF_W2P  3149824    // 512x1024  bf16 packed W2^T tiles       1 MiB
// total ~4.2 MiB

__device__ __forceinline__ unsigned short f2bf(float f) {
  unsigned int u = __float_as_uint(f);
  u += 0x7fffu + ((u >> 16) & 1u);      // round-to-nearest-even
  return (unsigned short)(u >> 16);
}

// m[p] = mean over N=8 of x[b, :, f],  p = b*4096 + f
__global__ void mean_kernel(const float* __restrict__ x, float* __restrict__ m) {
  int p = blockIdx.x * 256 + threadIdx.x;        // 262144 threads
  int b = p >> 12, f = p & 4095;
  const float* xp = x + (size_t)b * 32768 + f;
  float s = 0.f;
#pragma unroll
  for (int n = 0; n < 8; ++n) s += xp[n * 4096];
  m[p] = s * 0.125f;
}

// s0[o] = sum_c W0[o, c]
__global__ void s0_kernel(const float* __restrict__ W0, float* __restrict__ s0) {
  int o = blockIdx.x * 256 + threadIdx.x;        // 1024 threads
  float s = 0.f;
#pragma unroll
  for (int c = 0; c < 8; ++c) s += W0[o * 8 + c];
  s0[o] = s;
}

// Pack W^T into WMMA B-fragment order: tiles of 32(K) x 16(N) bf16.
// Packed idx = ((kt*NT + nt)*32 + lane)*16 + e
// with N = nt*16 + lane%16,  K = kt*32 + (lane/16)*8 + (e/8)*16 + (e%8)
__global__ void pack_w1(const float* __restrict__ W, unsigned short* __restrict__ P) {
  int idx = blockIdx.x * 256 + threadIdx.x;      // 1024*1024 threads
  int e = idx & 15, l = (idx >> 4) & 31, tile = idx >> 9;
  int nt = tile & 63, kt = tile >> 6;            // NT = 64
  int n = nt * 16 + (l & 15);
  int k = kt * 32 + (l >> 4) * 8 + ((e >> 3) << 4) + (e & 7);
  P[idx] = f2bf(W[n * 1024 + k]);                // W1 is (1024 x 1024) row-major
}

__global__ void pack_w2(const float* __restrict__ W, unsigned short* __restrict__ P) {
  int idx = blockIdx.x * 256 + threadIdx.x;      // 512*1024 threads
  int e = idx & 15, l = (idx >> 4) & 31, tile = idx >> 9;
  int nt = tile & 31, kt = tile >> 5;            // NT = 32
  int n = nt * 16 + (l & 15);
  int k = kt * 32 + (l >> 4) * 8 + ((e >> 3) << 4) + (e & 7);
  P[idx] = f2bf(W[n * 1024 + k]);                // W2 is (512 x 1024) row-major
}

// Fused: h1 tile in LDS -> GEMM1(W1)+relu -> h2 in LDS -> GEMM2(W2)+bias -> out
// 64 pixels per workgroup, 8 wave32s.
// Each wave owns a UNIQUE column eighth and all 4 rowblocks: B fragments are
// read from L2 exactly once per workgroup.
__global__ __launch_bounds__(256, 1) void fused_mlp(
    const float* __restrict__ b1g, const float* __restrict__ b2g,
    const float* __restrict__ b0g, const float* __restrict__ mg,
    const float* __restrict__ s0g, const unsigned short* __restrict__ W1p,
    const unsigned short* __restrict__ W2p, float* __restrict__ out) {
  extern __shared__ unsigned char smem[];
  unsigned short* s_h1 = (unsigned short*)smem;            // 64*1024 bf16 = 131072 B
  unsigned short* s_h2 = (unsigned short*)(smem + 131072); // 131072 B
  float* s_s0 = (float*)(smem + 262144);                   // 4 KiB
  float* s_b0 = (float*)(smem + 266240);                   // 4 KiB
  float* s_m  = (float*)(smem + 270336);                   // 256 B

  const int tid  = threadIdx.x;
  const int wave = tid >> 5, lane = tid & 31;
  const int lhi  = lane >> 4, llo = lane & 15;
  const int ptile = blockIdx.x;                  // rows [ptile*64, +64)

  for (int i = tid; i < 1024; i += 256) { s_s0[i] = s0g[i]; s_b0[i] = b0g[i]; }
  if (tid < 64) s_m[tid] = mg[ptile * 64 + tid];
  __syncthreads();

  // ---- phase 1: materialize h1 = relu(m*s0 + b0) once, bf16, row-major ----
  for (int c = tid; c < 8192; c += 256) {        // 8192 chunks of 8 elems
    int row = c >> 7;
    int k0  = (c & 127) << 3;
    float mv = s_m[row];
    union { uint4 q; unsigned short s[8]; } u;
#pragma unroll
    for (int j = 0; j < 8; ++j)
      u.s[j] = f2bf(fmaxf(mv * s_s0[k0 + j] + s_b0[k0 + j], 0.f));
    *(uint4*)(s_h1 + row * 1024 + k0) = u.q;     // ds_store_b128
  }
  __syncthreads();

  const v8f vzero = {0.f, 0.f, 0.f, 0.f, 0.f, 0.f, 0.f, 0.f};

  // ---------------- GEMM1: h2 = relu(h1 @ W1^T + b1) ----------------
  for (int g = 0; g < 4; ++g) {
    const int cbase = wave * 128 + g * 32;       // 2 col tiles per group
    v8f acc[4][2];
#pragma unroll
    for (int i = 0; i < 4; ++i)
#pragma unroll
      for (int t = 0; t < 2; ++t) acc[i][t] = vzero;

    for (int kt = 0; kt < 32; ++kt) {
      const int kbase = kt * 32 + lhi * 8;
      union { v16bf v; uint4 q[2]; } a[4];
#pragma unroll
      for (int i = 0; i < 4; ++i) {
        int row = i * 16 + llo;
        a[i].q[0] = *(const uint4*)(s_h1 + row * 1024 + kbase);
        a[i].q[1] = *(const uint4*)(s_h1 + row * 1024 + kbase + 16);
      }
#pragma unroll
      for (int t = 0; t < 2; ++t) {
        const v16bf bv = *(const v16bf*)(W1p +
            (size_t)(kt * 64 + (cbase >> 4) + t) * 512 + lane * 16);
#pragma unroll
        for (int i = 0; i < 4; ++i)
          acc[i][t] = __builtin_amdgcn_wmma_f32_16x16x32_bf16(
              false, a[i].v, false, bv, (short)0, acc[i][t], false, false);
      }
    }
    // epilogue: bias + relu -> bf16 LDS (C/D: VGPR r holds M = r + lhi*8)
#pragma unroll
    for (int t = 0; t < 2; ++t) {
      int col = cbase + t * 16 + llo;
      float bb = b1g[col];
#pragma unroll
      for (int i = 0; i < 4; ++i)
#pragma unroll
        for (int r = 0; r < 8; ++r) {
          int row = i * 16 + r + lhi * 8;
          s_h2[row * 1024 + col] = f2bf(fmaxf(acc[i][t][r] + bb, 0.f));
        }
    }
  }
  __syncthreads();

  // ---------------- GEMM2: out = h2 @ W2^T + b2 ----------------
  for (int g = 0; g < 2; ++g) {
    const int cbase = wave * 64 + g * 32;
    v8f acc[4][2];
#pragma unroll
    for (int i = 0; i < 4; ++i)
#pragma unroll
      for (int t = 0; t < 2; ++t) acc[i][t] = vzero;

    for (int kt = 0; kt < 32; ++kt) {
      const int kbase = kt * 32 + lhi * 8;
      union { v16bf v; uint4 q[2]; } a[4];
#pragma unroll
      for (int i = 0; i < 4; ++i) {
        int row = i * 16 + llo;
        a[i].q[0] = *(const uint4*)(s_h2 + row * 1024 + kbase);
        a[i].q[1] = *(const uint4*)(s_h2 + row * 1024 + kbase + 16);
      }
#pragma unroll
      for (int t = 0; t < 2; ++t) {
        const v16bf bv = *(const v16bf*)(W2p +
            (size_t)(kt * 32 + (cbase >> 4) + t) * 512 + lane * 16);
#pragma unroll
        for (int i = 0; i < 4; ++i)
          acc[i][t] = __builtin_amdgcn_wmma_f32_16x16x32_bf16(
              false, a[i].v, false, bv, (short)0, acc[i][t], false, false);
      }
    }
    // epilogue: bias + vectorized store (consecutive r -> consecutive f)
#pragma unroll
    for (int t = 0; t < 2; ++t) {
      int o = cbase + t * 16 + llo;
      float bb = b2g[o];
#pragma unroll
      for (int i = 0; i < 4; ++i) {
        int p0 = ptile * 64 + i * 16 + lhi * 8;            // rows p0..p0+7
        int bidx = p0 >> 12, f0 = p0 & 4095;               // same b for all 8
        float* dst = out + (size_t)bidx * 2097152 + (size_t)o * 4096 + f0;
        float4 lo = {acc[i][t][0] + bb, acc[i][t][1] + bb,
                     acc[i][t][2] + bb, acc[i][t][3] + bb};
        float4 hi = {acc[i][t][4] + bb, acc[i][t][5] + bb,
                     acc[i][t][6] + bb, acc[i][t][7] + bb};
        *(float4*)dst       = lo;                          // global_store_b128
        *(float4*)(dst + 4) = hi;
      }
    }
  }
}

extern "C" void kernel_launch(void* const* d_in, const int* in_sizes, int n_in,
                              void* d_out, int out_size, void* d_ws, size_t ws_size,
                              hipStream_t stream) {
  const float* x  = (const float*)d_in[0];
  const float* W0 = (const float*)d_in[1];
  const float* b0 = (const float*)d_in[2];
  const float* W1 = (const float*)d_in[3];
  const float* b1 = (const float*)d_in[4];
  const float* W2 = (const float*)d_in[5];
  const float* b2 = (const float*)d_in[6];
  float* out = (float*)d_out;

  unsigned char* ws = (unsigned char*)d_ws;
  float* m            = (float*)(ws + OFF_M);
  float* s0           = (float*)(ws + OFF_S0);
  unsigned short* W1p = (unsigned short*)(ws + OFF_W1P);
  unsigned short* W2p = (unsigned short*)(ws + OFF_W2P);

  mean_kernel<<<1024, 256, 0, stream>>>(x, m);       // 262144 pixels
  s0_kernel<<<4, 256, 0, stream>>>(W0, s0);          // 1024 sums
  pack_w1<<<4096, 256, 0, stream>>>(W1, W1p);        // 1M bf16
  pack_w2<<<2048, 256, 0, stream>>>(W2, W2p);        // 512K bf16

  const size_t smem = 131072 + 131072 + 4096 + 4096 + 256; // 270592 B (< 320 KB/WGP)
  (void)hipFuncSetAttribute((const void*)fused_mlp,
                            hipFuncAttributeMaxDynamicSharedMemorySize, (int)smem);
  fused_mlp<<<4096, 256, smem, stream>>>(b1, b2, b0, m, s0, W1p, W2p, out);
}
